// MambaPositionPredictor_83030307766882
// MI455X (gfx1250) — matrix operand order repaired and checked by reading, not compiled, not run on hardware
//
#include <hip/hip_runtime.h>
#include <hip/hip_bf16.h>

typedef __attribute__((ext_vector_type(16))) _Float16 v16h;
typedef __attribute__((ext_vector_type(8)))  float    v8f;

#define Bv      64
#define Tv      512
#define HID     288
#define NCHUNK  31      // 31 chunks * 16 steps = 496 = Tv - 16 (prefix before keep window)

// ---------------------------------------------------------------------------
// Encode: sinusoidal spatial features + sp proj + input proj -> h0 (B,T,96)
// ---------------------------------------------------------------------------
__global__ __launch_bounds__(256) void encode_kernel(
    const float* __restrict__ x, const int* __restrict__ lengths,
    const float* __restrict__ sp_w, const float* __restrict__ sp_b,
    const float* __restrict__ in_w, const float* __restrict__ in_b,
    float* __restrict__ h0)
{
    __shared__ float sw[5632]; // sp_w 2048 | sp_b 32 | in_w 3456 | in_b 96
    for (int i = threadIdx.x; i < 5632; i += 256) {
        float v;
        if      (i < 2048) v = sp_w[i];
        else if (i < 2080) v = sp_b[i - 2048];
        else if (i < 5536) v = in_w[i - 2080];
        else               v = in_b[i - 5536];
        sw[i] = v;
    }
    __syncthreads();
    const float* s_spw = sw;
    const float* s_spb = sw + 2048;
    const float* s_inw = sw + 2080;
    const float* s_inb = sw + 5536;

    int tok = blockIdx.x * 256 + threadIdx.x;
    int b = tok / Tv, t = tok - b * Tv;
    float msk = (t < lengths[b]) ? 1.f : 0.f;
    float xi[4];
#pragma unroll
    for (int c = 0; c < 4; ++c) xi[c] = x[(size_t)tok * 4 + c];

    float enc[64];
#pragma unroll
    for (int i = 0; i < 8; ++i) {
        // 1/freq_i = 10000^(i/4) = 2^(log2(10000)/4 * i)
        float invf = exp2f(3.3219280949f * (float)i);
#pragma unroll
        for (int c = 0; c < 4; ++c) {
            float ang = xi[c] * invf;
            enc[c * 16 + i]     = __sinf(ang) * msk;
            enc[c * 16 + 8 + i] = __cosf(ang) * msk;
        }
    }
    float pin[36];
#pragma unroll
    for (int c = 0; c < 4; ++c) pin[c] = xi[c];
    for (int o = 0; o < 32; ++o) {
        float a = s_spb[o];
#pragma unroll 8
        for (int e = 0; e < 64; ++e) a = __builtin_fmaf(s_spw[o * 64 + e], enc[e], a);
        pin[4 + o] = a;
    }
    for (int mo = 0; mo < 96; ++mo) {
        float a = s_inb[mo];
#pragma unroll 6
        for (int p = 0; p < 36; ++p) a = __builtin_fmaf(s_inw[mo * 36 + p], pin[p], a);
        h0[(size_t)tok * 96 + mo] = a;
    }
}

// ---------------------------------------------------------------------------
// Generic WMMA GEMM:  C[r, n] = act( A[map(r), k] * W[n, k]^T + bias[n] )
// A rows map through (b = r/rows_per_b, t = t0 + r%rows_per_b, optional flip)
// so the time-flipped mamba never materializes a flipped h0.
// 4 waves/block, each wave owns one 16-col tile, A tile shared via LDS.
// ---------------------------------------------------------------------------
__global__ __launch_bounds__(128) void gemm_wmma_kernel(
    const float* __restrict__ A, int lda, int A_T, int rows_per_b, int t0, int flipT,
    const float* __restrict__ W, const float* __restrict__ bias,
    float* __restrict__ C, int ldc, int C_T,
    int M, int N, int K, int act)
{
    extern __shared__ _Float16 lds[];
    const int Kpad = (K + 31) & ~31;
    _Float16* aT = lds;              // [16][Kpad]
    _Float16* wT = lds + 16 * Kpad;  // [64][Kpad]
    const int m0 = blockIdx.y * 16;
    const int n0 = blockIdx.x * 64;
    const int tid = threadIdx.x;

    for (int idx = tid; idx < 16 * Kpad; idx += 128) {
        int r = idx / Kpad, k = idx - r * Kpad;
        int mrow = m0 + r;
        float v = 0.f;
        if (k < K && mrow < M) {
            int bb = mrow / rows_per_b;
            int tt = t0 + (mrow - bb * rows_per_b);
            if (flipT) tt = flipT - 1 - tt;
            v = A[((size_t)bb * A_T + tt) * lda + k];
        }
        aT[idx] = (_Float16)v;
    }
    for (int idx = tid; idx < 64 * Kpad; idx += 128) {
        int nl = idx / Kpad, k = idx - nl * Kpad;
        int n = n0 + nl;
        float v = (k < K && n < N) ? W[(size_t)n * K + k] : 0.f;
        wT[idx] = (_Float16)v;
    }
    __syncthreads();

    const int wave = tid >> 5, lane = tid & 31;
    const bool lo = lane < 16;
    const int mr = lane & 15;
    const _Float16* arow = aT + mr * Kpad;
    const _Float16* wrow = wT + (wave * 16 + mr) * Kpad;

    v8f acc = {};
    for (int kk = 0; kk < Kpad; kk += 32) {
        v16h va, vb;
        // A 16-bit 16x32 layout: lanes 0-15 -> K {0..7,16..23}, lanes 16-31 -> K {8..15,24..31}
        int ka0 = kk + (lo ? 0 : 8);
        int ka1 = kk + (lo ? 16 : 24);
#pragma unroll
        for (int j = 0; j < 8; ++j) { va[j] = arow[ka0 + j]; va[8 + j] = arow[ka1 + j]; }
        // B 16-bit 32x16 layout: lanes 0-15 -> K {0..7,16..23}, lanes 16-31 -> K {8..15,24..31}, col = lane&15
        int kb0 = kk + (lo ? 0 : 8);
        int kb1 = kk + (lo ? 16 : 24);
#pragma unroll
        for (int j = 0; j < 8; ++j) { vb[j] = wrow[kb0 + j]; vb[8 + j] = wrow[kb1 + j]; }
        acc = __builtin_amdgcn_wmma_f32_16x16x32_f16(
            false, va, false, vb, (short)0, acc, false, false);
    }

    int ncol = n0 + wave * 16 + mr;
    if (ncol < N) {
        float bvv = bias ? bias[ncol] : 0.f;
#pragma unroll
        for (int i = 0; i < 8; ++i) {
            int r = m0 + i + (lo ? 0 : 8);
            if (r < M) {
                int bb = r / rows_per_b;
                int tt = r - bb * rows_per_b;
                float v = acc[i] + bvv;
                if      (act == 1) v = tanhf(v);
                else if (act == 2) v = fmaxf(v, 0.f);
                C[((size_t)bb * C_T + tt) * ldc + ncol] = v;
            }
        }
    }
}

// ---------------------------------------------------------------------------
// Depthwise causal conv(4) + SiLU, then x-proj (22) and dt-proj + softplus.
// One block per (b,t) token in logical scan time; 192 threads = channels.
// ---------------------------------------------------------------------------
__global__ __launch_bounds__(192) void conv_xproj_kernel(
    const float* __restrict__ xraw, const float* __restrict__ conv_w,
    const float* __restrict__ conv_b, const float* __restrict__ xproj_w,
    const float* __restrict__ dt_w, const float* __restrict__ dt_b,
    float* __restrict__ xact, float* __restrict__ dtf, float* __restrict__ bcb,
    int Tcomp)
{
    int blk = blockIdx.x;
    int b = blk / Tcomp, t = blk - b * Tcomp;
    int d = threadIdx.x;
    size_t tok = (size_t)b * Tv + t;
    __shared__ float sx[192];
    __shared__ float sdbc[22];

    float acc = conv_b[d];
#pragma unroll
    for (int k = 0; k < 4; ++k) {
        int tt = t - 3 + k;
        float xv = (tt >= 0) ? xraw[((size_t)b * Tv + tt) * 192 + d] : 0.f;
        acc = __builtin_fmaf(xv, conv_w[d * 4 + k], acc);
    }
    float xa = acc / (1.f + __expf(-acc)); // SiLU
    xact[tok * 192 + d] = xa;
    sx[d] = xa;
    __syncthreads();
    if (d < 22) {
        float s = 0.f;
        for (int j = 0; j < 192; ++j) s = __builtin_fmaf(xproj_w[d * 192 + j], sx[j], s);
        sdbc[d] = s;
    }
    __syncthreads();
    float sdt = dt_b[d];
#pragma unroll
    for (int r = 0; r < 6; ++r) sdt = __builtin_fmaf(dt_w[d * 6 + r], sdbc[r], sdt);
    float dtv = (sdt > 20.f) ? sdt : __logf(1.f + __expf(sdt)); // softplus
    dtf[tok * 192 + d] = dtv;
    if (d < 16) bcb[tok * 16 + d] = sdbc[6 + d]; // B(8) then C(8)
}

// ---------------------------------------------------------------------------
// Short serial scan (used for the causal forward branch: only 16 steps).
// ---------------------------------------------------------------------------
__global__ __launch_bounds__(192) void scan_kernel(
    const float* __restrict__ xact, const float* __restrict__ dtf,
    const float* __restrict__ bcb, const float* __restrict__ zkeep,
    const float* __restrict__ A_log, const float* __restrict__ Dp,
    float* __restrict__ yg, int keep0, int t_end)
{
    int b = blockIdx.x, d = threadIdx.x;
    float An[8];
#pragma unroll
    for (int n = 0; n < 8; ++n) An[n] = -__expf(A_log[d * 8 + n]);
    float Dd = Dp[d];
    float s[8] = {0.f, 0.f, 0.f, 0.f, 0.f, 0.f, 0.f, 0.f};
    __shared__ float sbc[2][16];
    size_t base = (size_t)b * Tv;

    for (int t = 0; t < t_end; ++t) {
        int buf = t & 1;
        if (d < 16) sbc[buf][d] = bcb[(base + t) * 16 + d];
        __syncthreads();
        float dtv = dtf[(base + t) * 192 + d];
        float u   = xact[(base + t) * 192 + d];
        float du = dtv * u;
#pragma unroll
        for (int n = 0; n < 8; ++n) {
            float dA = __expf(dtv * An[n]);
            s[n] = __builtin_fmaf(dA, s[n], du * sbc[buf][n]);
        }
        int j = t - keep0;
        if (j >= 0 && j < 16) {
            float y = 0.f;
#pragma unroll
            for (int n = 0; n < 8; ++n) y = __builtin_fmaf(s[n], sbc[buf][8 + n], y);
            float z = zkeep[((size_t)b * 16 + j) * 192 + d];
            float sig = 1.f / (1.f + __expf(-z));
            yg[((size_t)b * 16 + j) * 192 + d] = (y + Dd * u) * z * sig;
        }
    }
}

// ---------------------------------------------------------------------------
// Chunked parallel scan, pass 1: each 16-step chunk reduces to an affine map
// s_out = Ach * s_in + Bch  per (b, chunk, d, n).  Fully parallel: B*NCHUNK
// blocks saturate the WGPs instead of one 512-step serial loop.
// ---------------------------------------------------------------------------
__global__ __launch_bounds__(192) void scan_chunk_kernel(
    const float* __restrict__ xact, const float* __restrict__ dtf,
    const float* __restrict__ bcb, const float* __restrict__ A_log,
    float* __restrict__ Ach, float* __restrict__ Bch)
{
    int b = blockIdx.x, c = blockIdx.y;
    int d = threadIdx.x;
    __shared__ float sB[16][8];
    size_t base = (size_t)b * Tv + (size_t)c * 16;
    for (int i = d; i < 128; i += 192) {
        int t = i >> 3, n = i & 7;
        sB[t][n] = bcb[(base + t) * 16 + n];
    }
    __syncthreads();
    float An[8];
#pragma unroll
    for (int n = 0; n < 8; ++n) An[n] = -__expf(A_log[d * 8 + n]);
    float a[8] = {1.f, 1.f, 1.f, 1.f, 1.f, 1.f, 1.f, 1.f};
    float bb[8] = {0.f, 0.f, 0.f, 0.f, 0.f, 0.f, 0.f, 0.f};
    for (int t = 0; t < 16; ++t) {
        float dtv = dtf[(base + t) * 192 + d];
        float u   = xact[(base + t) * 192 + d];
        float du = dtv * u;
#pragma unroll
        for (int n = 0; n < 8; ++n) {
            float dA = __expf(dtv * An[n]);
            a[n]  = a[n] * dA;
            bb[n] = __builtin_fmaf(dA, bb[n], du * sB[t][n]);
        }
    }
    size_t o = (((size_t)b * NCHUNK + c) * 192 + d) * 8;
#pragma unroll
    for (int n = 0; n < 8; ++n) { Ach[o + n] = a[n]; Bch[o + n] = bb[n]; }
}

// ---------------------------------------------------------------------------
// Chunked parallel scan, pass 2: compose the NCHUNK affine maps sequentially
// (short critical path), then run the real last-16 steps with z-gated output.
// ---------------------------------------------------------------------------
__global__ __launch_bounds__(192) void scan_tail_kernel(
    const float* __restrict__ xact, const float* __restrict__ dtf,
    const float* __restrict__ bcb, const float* __restrict__ zkeep,
    const float* __restrict__ A_log, const float* __restrict__ Dp,
    const float* __restrict__ Ach, const float* __restrict__ Bch,
    float* __restrict__ yg, int keep0)
{
    int b = blockIdx.x, d = threadIdx.x;
    float s[8] = {0.f, 0.f, 0.f, 0.f, 0.f, 0.f, 0.f, 0.f};
    for (int c = 0; c < NCHUNK; ++c) {
        size_t o = (((size_t)b * NCHUNK + c) * 192 + d) * 8;
#pragma unroll
        for (int n = 0; n < 8; ++n) s[n] = __builtin_fmaf(Ach[o + n], s[n], Bch[o + n]);
    }
    float An[8];
#pragma unroll
    for (int n = 0; n < 8; ++n) An[n] = -__expf(A_log[d * 8 + n]);
    float Dd = Dp[d];
    __shared__ float sbc[2][16];
    size_t base = (size_t)b * Tv;
    for (int j = 0; j < 16; ++j) {
        int t = keep0 + j;
        int buf = j & 1;
        if (d < 16) sbc[buf][d] = bcb[(base + t) * 16 + d];
        __syncthreads();
        float dtv = dtf[(base + t) * 192 + d];
        float u   = xact[(base + t) * 192 + d];
        float du = dtv * u;
        float y = 0.f;
#pragma unroll
        for (int n = 0; n < 8; ++n) {
            float dA = __expf(dtv * An[n]);
            s[n] = __builtin_fmaf(dA, s[n], du * sbc[buf][n]);
            y    = __builtin_fmaf(s[n], sbc[buf][8 + n], y);
        }
        float z = zkeep[((size_t)b * 16 + j) * 192 + d];
        float sig = 1.f / (1.f + __expf(-z));
        yg[((size_t)b * 16 + j) * 192 + d] = (y + Dd * u) * z * sig;
    }
}

// ---------------------------------------------------------------------------
// Concat [h_fwd, h_bwd(reversed keep index), h0] + LayerNorm over 288.
// ---------------------------------------------------------------------------
__global__ __launch_bounds__(HID) void ln_kernel(
    const float* __restrict__ h1, const float* __restrict__ h2,
    const float* __restrict__ h0, const float* __restrict__ g,
    const float* __restrict__ bb, float* __restrict__ out)
{
    int row = blockIdx.x;
    int b = row / 16, j = row - b * 16;
    int tid = threadIdx.x;
    float v;
    if      (tid < 96)  v = h1[((size_t)b * 16 + j) * 96 + tid];
    else if (tid < 192) v = h2[((size_t)b * 16 + (15 - j)) * 96 + (tid - 96)];
    else                v = h0[((size_t)b * Tv + j) * 96 + (tid - 192)];

    __shared__ float red[HID];
    __shared__ float red2[32];
    __shared__ float stats[2];
    red[tid] = v; __syncthreads();
    if (tid < 32) { float s = 0.f; for (int i = tid; i < HID; i += 32) s += red[i]; red2[tid] = s; }
    __syncthreads();
    if (tid == 0) { float s = 0.f; for (int i = 0; i < 32; ++i) s += red2[i]; stats[0] = s / HID; }
    __syncthreads();
    float mu = stats[0];
    float d0 = v - mu;
    red[tid] = d0 * d0; __syncthreads();
    if (tid < 32) { float s = 0.f; for (int i = tid; i < HID; i += 32) s += red[i]; red2[tid] = s; }
    __syncthreads();
    if (tid == 0) { float s = 0.f; for (int i = 0; i < 32; ++i) s += red2[i]; stats[1] = s / HID; }
    __syncthreads();
    float inv = rsqrtf(stats[1] + 1e-5f);
    out[(size_t)row * HID + tid] = d0 * inv * g[tid] + bb[tid];
}

// ---------------------------------------------------------------------------
// Final 72->2 heads + output slice: out[b, j, {xy0,xy1,wh0,wh1}], j < target_len
// ---------------------------------------------------------------------------
__global__ __launch_bounds__(256) void head3_kernel(
    const float* __restrict__ t2xy, const float* __restrict__ t2wh,
    const float* __restrict__ xy3w, const float* __restrict__ xy3b,
    const float* __restrict__ wh3w, const float* __restrict__ wh3b,
    float* __restrict__ out, int target_len, int total)
{
    int idx = blockIdx.x * 256 + threadIdx.x;
    if (idx >= total) return;
    int c = idx & 3;
    int j = (idx >> 2) % target_len;
    int b = idx / (4 * target_len);
    const float* src; const float* w; float bias;
    if (c < 2) { src = t2xy + ((size_t)b * 16 + j) * 72; w = xy3w + c * 72;       bias = xy3b[c]; }
    else       { src = t2wh + ((size_t)b * 16 + j) * 72; w = wh3w + (c - 2) * 72; bias = wh3b[c - 2]; }
    float s = bias;
    for (int i = 0; i < 72; ++i) s = __builtin_fmaf(w[i], src[i], s);
    out[idx] = s;
}

// ---------------------------------------------------------------------------
extern "C" void kernel_launch(void* const* d_in, const int* in_sizes, int n_in,
                              void* d_out, int out_size, void* d_ws, size_t ws_size,
                              hipStream_t stream)
{
    (void)in_sizes; (void)n_in; (void)ws_size;
    const float* x       = (const float*)d_in[0];
    const int*   lengths = (const int*)  d_in[1];
    const float* sp_w = (const float*)d_in[3];
    const float* sp_b = (const float*)d_in[4];
    const float* in_w = (const float*)d_in[5];
    const float* in_b = (const float*)d_in[6];
    const float* m_inw[2]   = {(const float*)d_in[7],  (const float*)d_in[16]};
    const float* m_convw[2] = {(const float*)d_in[8],  (const float*)d_in[17]};
    const float* m_convb[2] = {(const float*)d_in[9],  (const float*)d_in[18]};
    const float* m_xpw[2]   = {(const float*)d_in[10], (const float*)d_in[19]};
    const float* m_dtw[2]   = {(const float*)d_in[11], (const float*)d_in[20]};
    const float* m_dtb[2]   = {(const float*)d_in[12], (const float*)d_in[21]};
    const float* m_alog[2]  = {(const float*)d_in[13], (const float*)d_in[22]};
    const float* m_D[2]     = {(const float*)d_in[14], (const float*)d_in[23]};
    const float* m_outw[2]  = {(const float*)d_in[15], (const float*)d_in[24]};
    const float* ln_g = (const float*)d_in[25];
    const float* ln_b = (const float*)d_in[26];
    const float* xy1w = (const float*)d_in[27]; const float* xy1b = (const float*)d_in[28];
    const float* xy2w = (const float*)d_in[29]; const float* xy2b = (const float*)d_in[30];
    const float* xy3w = (const float*)d_in[31]; const float* xy3b = (const float*)d_in[32];
    const float* wh1w = (const float*)d_in[33]; const float* wh1b = (const float*)d_in[34];
    const float* wh2w = (const float*)d_in[35]; const float* wh2b = (const float*)d_in[36];
    const float* wh3w = (const float*)d_in[37]; const float* wh3b = (const float*)d_in[38];

    float* ws    = (float*)d_ws;
    float* h0v   = ws;                                   // B*T*96
    float* xraw  = h0v   + (size_t)Bv * Tv * 96;         // B*T*192
    float* xactv = xraw  + (size_t)Bv * Tv * 192;        // B*T*192
    float* dtfv  = xactv + (size_t)Bv * Tv * 192;        // B*T*192
    float* bcv   = dtfv  + (size_t)Bv * Tv * 192;        // B*T*16
    float* zkeep = bcv   + (size_t)Bv * Tv * 16;         // B*16*192
    float* yg1   = zkeep + (size_t)Bv * 16 * 192;        // B*16*192
    float* yg2   = yg1   + (size_t)Bv * 16 * 192;        // B*16*192
    float* h1v   = yg2   + (size_t)Bv * 16 * 192;        // B*16*96
    float* h2v   = h1v   + (size_t)Bv * 16 * 96;         // B*16*96
    float* lnout = h2v   + (size_t)Bv * 16 * 96;         // B*16*288
    float* t1xy  = lnout + (size_t)Bv * 16 * 288;        // B*16*144
    float* t1wh  = t1xy  + (size_t)Bv * 16 * 144;        // B*16*144
    float* t2xy  = t1wh  + (size_t)Bv * 16 * 144;        // B*16*72
    float* t2wh  = t2xy  + (size_t)Bv * 16 * 72;         // B*16*72
    float* Ach   = t2wh  + (size_t)Bv * 16 * 72;         // B*NCHUNK*192*8
    float* Bch   = Ach   + (size_t)Bv * NCHUNK * 192 * 8;

    int target_len = out_size / (Bv * 4);

    encode_kernel<<<(Bv * Tv) / 256, 256, 0, stream>>>(x, lengths, sp_w, sp_b, in_w, in_b, h0v);

    auto gemm = [&](const float* A, int lda, int A_T, int rows_per_b, int t0, int flipT,
                    const float* W, const float* bias, float* C, int ldc, int C_T,
                    int M, int N, int K, int act) {
        int Kpad = (K + 31) & ~31;
        dim3 grid((N + 63) / 64, M / 16);
        size_t sh = (size_t)(16 + 64) * Kpad * sizeof(_Float16);
        gemm_wmma_kernel<<<grid, 128, sh, stream>>>(A, lda, A_T, rows_per_b, t0, flipT,
                                                    W, bias, C, ldc, C_T, M, N, K, act);
    };

    for (int mi = 0; mi < 2; ++mi) {
        int flip  = mi ? Tv : 0;           // mamba2 reads h0 time-flipped
        int keep0 = mi ? (Tv - 16) : 0;    // keep window in logical scan time
        int Mx    = mi ? (Bv * Tv) : (Bv * 16);  // fwd branch: only first 16 tokens matter
        int rpbx  = mi ? Tv : 16;
        int Tcomp = mi ? Tv : 16;
        // in-proj x-half (conv input), logical time, full/restricted T
        gemm(h0v, 96, Tv, rpbx, 0, flip, m_inw[mi], nullptr,
             xraw, 192, Tv, Mx, 192, 96, 0);
        // in-proj z-half only on the keep window
        gemm(h0v, 96, Tv, 16, keep0, flip, m_inw[mi] + 192 * 96, nullptr,
             zkeep, 192, 16, Bv * 16, 192, 96, 0);
        conv_xproj_kernel<<<Bv * Tcomp, 192, 0, stream>>>(
            xraw, m_convw[mi], m_convb[mi], m_xpw[mi], m_dtw[mi], m_dtb[mi],
            xactv, dtfv, bcv, Tcomp);
        float* yg = mi ? yg2 : yg1;
        if (mi == 0) {
            // causal branch: only 16 real steps
            scan_kernel<<<Bv, 192, 0, stream>>>(xactv, dtfv, bcv, zkeep,
                                                m_alog[mi], m_D[mi], yg, 0, 16);
        } else {
            // backward branch: chunked parallel scan over the 496-step prefix,
            // then sequential combine + final 16 gated steps.
            scan_chunk_kernel<<<dim3(Bv, NCHUNK), 192, 0, stream>>>(
                xactv, dtfv, bcv, m_alog[mi], Ach, Bch);
            scan_tail_kernel<<<Bv, 192, 0, stream>>>(
                xactv, dtfv, bcv, zkeep, m_alog[mi], m_D[mi], Ach, Bch, yg, keep0);
        }
        gemm(yg, 192, 16, 16, 0, 0, m_outw[mi], nullptr,
             mi ? h2v : h1v, 96, 16, Bv * 16, 96, 192, 0);
    }

    ln_kernel<<<Bv * 16, HID, 0, stream>>>(h1v, h2v, h0v, ln_g, ln_b, lnout);

    gemm(lnout, 288, 16, 16, 0, 0, xy1w, xy1b, t1xy, 144, 16, Bv * 16, 144, 288, 1);
    gemm(lnout, 288, 16, 16, 0, 0, wh1w, wh1b, t1wh, 144, 16, Bv * 16, 144, 288, 1);
    gemm(t1xy, 144, 16, 16, 0, 0, xy2w, xy2b, t2xy, 72, 16, Bv * 16, 72, 144, 2);
    gemm(t1wh, 144, 16, 16, 0, 0, wh2w, wh2b, t2wh, 72, 16, Bv * 16, 72, 144, 2);

    int total = Bv * target_len * 4;
    head3_kernel<<<(total + 255) / 256, 256, 0, stream>>>(
        t2xy, t2wh, xy3w, xy3b, wh3w, wh3b, (float*)d_out, target_len, total);
}
